// DeformConv2D_25048249270337
// MI455X (gfx1250) — compile-verified, matrix operand order checked by reading, not run on hardware
//
#include <hip/hip_runtime.h>
#include <hip/hip_bf16.h>

typedef __attribute__((ext_vector_type(16))) __bf16 v16bf;
typedef __attribute__((ext_vector_type(8)))  __bf16 v8bf;
typedef __attribute__((ext_vector_type(8)))  float  v8f;

#define HH    128
#define WW    128
#define HWSZ  16384
#define CIN   64
#define COUT  128
#define BATCH 8
#define KKK   9
#define KDIM  576          // CIN * 9, contraction depth
#define NT    32           // pixels per workgroup tile
#define SROW  584          // padded LDS K stride (mult of 8 elems -> 16B rows; 36-dword stride -> conflict-free b128 frags)

// ---------------- Kernel A0: weight fp32 -> bf16 [COUT][KDIM] ----------------
__global__ void wconv_bf16_kernel(const float* __restrict__ w, __bf16* __restrict__ wb) {
    int i = blockIdx.x * 256 + threadIdx.x;
    if (i < COUT * KDIM) wb[i] = (__bf16)w[i];
}

// ---------------- Kernel A1: offset = conv3x3(x, offset_w) + offset_b --------
__global__ __launch_bounds__(256) void offconv_kernel(const float* __restrict__ x,
                                                      const float* __restrict__ ow,
                                                      const float* __restrict__ ob,
                                                      float* __restrict__ offs) {
    __shared__ float sow[18 * KDIM];
    __shared__ float sob[18];
    for (int i = threadIdx.x; i < 18 * KDIM; i += 256) sow[i] = ow[i];
    if (threadIdx.x < 18) sob[threadIdx.x] = ob[threadIdx.x];
    __syncthreads();

    int t   = blockIdx.x * 256 + threadIdx.x;   // one thread per (b, pixel)
    int b   = t >> 14;
    int pix = t & (HWSZ - 1);
    int y   = pix >> 7;
    int xc  = pix & 127;

    float acc[18];
#pragma unroll
    for (int ch = 0; ch < 18; ++ch) acc[ch] = sob[ch];

    const float* xb = x + (size_t)b * CIN * HWSZ;
    for (int pos = 0; pos < 9; ++pos) {
        int yy = y + pos / 3 - 1;
        int xx = xc + pos % 3 - 1;
        if (yy < 0 || yy >= HH || xx < 0 || xx >= WW) continue;   // zero padding
        const float* xp = xb + yy * WW + xx;
        for (int c = 0; c < CIN; ++c) {
            float xv = xp[c * HWSZ];                 // coalesced across lanes
            const float* wp = sow + c * 9 + pos;     // wave-uniform LDS broadcast
#pragma unroll
            for (int ch = 0; ch < 18; ++ch) acc[ch] += xv * wp[ch * KDIM];
        }
    }
    float* op = offs + (size_t)(b * 18) * HWSZ + pix;
#pragma unroll
    for (int ch = 0; ch < 18; ++ch) op[ch * HWSZ] = acc[ch];
}

// ------- Kernel B: fused bilinear sampling (-> LDS bf16) + WMMA GEMM --------
__global__ __launch_bounds__(256) void deform_wmma_kernel(const float* __restrict__ x,
                                                          const __bf16* __restrict__ wb,
                                                          const float* __restrict__ bias,
                                                          const float* __restrict__ offs,
                                                          float* __restrict__ out) {
    __shared__ __bf16 sS[NT * SROW];   // S tile: [pixel][k], k = c*9+kk, padded rows

    int b    = blockIdx.x >> 9;        // 512 pixel-tiles per batch
    int tile = blockIdx.x & 511;
    int p0   = tile * NT;
    const float* xb = x + (size_t)b * CIN * HWSZ;

    // ---- phase 1: compute sampled[c, kk, p] for this 32-pixel tile ----
    for (int idx = threadIdx.x; idx < NT * KKK; idx += 256) {
        int pp  = idx / 9;
        int kk  = idx - pp * 9;
        int pix = p0 + pp;
        int y   = pix >> 7;
        int xc  = pix & 127;

        float dy = offs[(size_t)(b * 18 + 2 * kk) * HWSZ + pix];
        float dx = offs[(size_t)(b * 18 + 2 * kk + 1) * HWSZ + pix];
        float py = (float)(y + kk / 3 - 1) + dy;
        float px = (float)(xc + kk % 3 - 1) + dx;

        float fy = floorf(py), fx = floorf(px);
        int iy0 = (int)fy, ix0 = (int)fx;
        int iy1 = iy0 + 1, ix1 = ix0 + 1;
        float wy1 = py - fy, wy0 = 1.0f - wy1;
        float wx1 = px - fx, wx0 = 1.0f - wx1;

        float vy0 = (iy0 >= 0 && iy0 < HH) ? 1.0f : 0.0f;
        float vy1 = (iy1 >= 0 && iy1 < HH) ? 1.0f : 0.0f;
        float vx0 = (ix0 >= 0 && ix0 < WW) ? 1.0f : 0.0f;
        float vx1 = (ix1 >= 0 && ix1 < WW) ? 1.0f : 0.0f;
        int cy0 = min(max(iy0, 0), HH - 1), cy1 = min(max(iy1, 0), HH - 1);
        int cx0 = min(max(ix0, 0), WW - 1), cx1 = min(max(ix1, 0), WW - 1);

        float w00 = wy0 * wx0 * vy0 * vx0;
        float w01 = wy0 * wx1 * vy0 * vx1;
        float w10 = wy1 * wx0 * vy1 * vx0;
        float w11 = wy1 * wx1 * vy1 * vx1;
        int l00 = cy0 * WW + cx0, l01 = cy0 * WW + cx1;
        int l10 = cy1 * WW + cx0, l11 = cy1 * WW + cx1;

        __bf16* srow = sS + pp * SROW + kk;   // indices/weights reused over 64 channels
#pragma unroll 4
        for (int c = 0; c < CIN; ++c) {
            const float* xp = xb + c * HWSZ;
            float v = w00 * xp[l00] + w01 * xp[l01] + w10 * xp[l10] + w11 * xp[l11];
            srow[c * 9] = (__bf16)v;
        }
    }
    __syncthreads();

    // ---- phase 2: D[128,32] += W[128,576] x S[576,32] via bf16 WMMA ----
    int lane = threadIdx.x & 31;
    int wave = threadIdx.x >> 5;          // 8 waves = 8 M-tiles (all of Cout)
    int col  = lane & 15;                 // row (A) / column (B) index
    int sel8 = (lane >> 4) << 3;          // 16-bit A/B layout: lane-half picks 8-elem K chunk

    v8f acc0 = {};
    v8f acc1 = {};
    const __bf16* arow  = wb + (size_t)(wave * 16 + col) * KDIM + sel8;
    const __bf16* brow0 = sS + col * SROW + sel8;
    const __bf16* brow1 = sS + (16 + col) * SROW + sel8;

    for (int kb = 0; kb < 18; ++kb) {
        int k0 = kb * 32;
        v8bf a0  = *(const v8bf*)(arow + k0);
        v8bf a1  = *(const v8bf*)(arow + k0 + 16);
        v8bf b00 = *(const v8bf*)(brow0 + k0);
        v8bf b01 = *(const v8bf*)(brow0 + k0 + 16);
        v8bf b10 = *(const v8bf*)(brow1 + k0);
        v8bf b11 = *(const v8bf*)(brow1 + k0 + 16);
        v16bf a, b0, b1;
#pragma unroll
        for (int i = 0; i < 8; ++i) {
            a[i] = a0[i];  a[i + 8] = a1[i];
            b0[i] = b00[i]; b0[i + 8] = b01[i];
            b1[i] = b10[i]; b1[i + 8] = b11[i];
        }
        acc0 = __builtin_amdgcn_wmma_f32_16x16x32_bf16(false, a, false, b0, (short)0, acc0, false, false);
        acc1 = __builtin_amdgcn_wmma_f32_16x16x32_bf16(false, a, false, b1, (short)0, acc1, false, false);
    }

    // ---- epilogue: C/D layout -> lanes 0-15: M=r, lanes 16-31: M=r+8 ----
    int mbase = wave * 16 + ((lane >> 4) << 3);
    float* ob = out + (size_t)(b * COUT) * HWSZ + p0;
#pragma unroll
    for (int r = 0; r < 8; ++r) {
        int m = mbase + r;
        float bv = bias[m];
        ob[(size_t)m * HWSZ + col]      = acc0[r] + bv;
        ob[(size_t)m * HWSZ + 16 + col] = acc1[r] + bv;
    }
}

extern "C" void kernel_launch(void* const* d_in, const int* in_sizes, int n_in,
                              void* d_out, int out_size, void* d_ws, size_t ws_size,
                              hipStream_t stream) {
    (void)in_sizes; (void)n_in; (void)out_size; (void)ws_size;
    const float* x        = (const float*)d_in[0];
    const float* weight   = (const float*)d_in[1];
    const float* bias     = (const float*)d_in[2];
    const float* offset_w = (const float*)d_in[3];
    const float* offset_b = (const float*)d_in[4];
    float* out = (float*)d_out;

    // workspace layout: [offsets fp32: 8*18*16384] [weight bf16: 128*576]
    float*  offs = (float*)d_ws;
    __bf16* wbuf = (__bf16*)((char*)d_ws + (size_t)BATCH * 18 * HWSZ * sizeof(float));

    wconv_bf16_kernel<<<(COUT * KDIM + 255) / 256, 256, 0, stream>>>(weight, wbuf);
    offconv_kernel<<<(BATCH * HWSZ) / 256, 256, 0, stream>>>(x, offset_w, offset_b, offs);
    deform_wmma_kernel<<<BATCH * 512, 256, 0, stream>>>(x, wbuf, bias, offs, out);
}
